// Attention_90666759618941
// MI455X (gfx1250) — matrix operand order repaired and checked by reading, not compiled, not run on hardware
//
#include <hip/hip_runtime.h>
#include <hip/hip_bf16.h>

typedef _Float16 half_t;
typedef __attribute__((ext_vector_type(16))) _Float16 v16h;
typedef __attribute__((ext_vector_type(8)))  _Float16 v8h;
typedef __attribute__((ext_vector_type(4)))  _Float16 v4h;
typedef __attribute__((ext_vector_type(8)))  float    v8f;

#define D_MODEL 1024
#define SEQ     2048
#define BATCH   4
#define NH      16
#define DH      64
#define MTOT    (BATCH * SEQ)   // 8192

#define BM 256                  // rows per block (8 waves x 32 rows)
#define BN 64                   // cols per block
#define BK 64                   // k per stage
#define LDAS (BK + 8)           // padded LDS row stride (halfs): 144B, 16B-aligned

// ---------------- WMMA fragment helpers (layouts per ISA 7.12.2) -------------

__device__ __forceinline__ v8f wmma_f16(v16h a, v16h b, v8f c) {
    return __builtin_amdgcn_wmma_f32_16x16x32_f16(false, a, false, b,
                                                  (short)0, c, false, false);
}

// A fragment: 16x32 (M x K), row-major source, leading dim `lda` (halfs).
__device__ __forceinline__ v16h load_a_frag(const half_t* base, int lda, int lane) {
    int m  = lane & 15;
    int hf = lane >> 4;
    const half_t* p = base + (size_t)m * lda;
    v8h lo = *(const v8h*)(p + hf * 8);
    v8h hi = *(const v8h*)(p + 16 + hf * 8);
    v16h a;
#pragma unroll
    for (int i = 0; i < 8; ++i) { a[i] = lo[i]; a[i + 8] = hi[i]; }
    return a;
}

// B fragment: 32x16 (K x N). Source stored N-major: row n holds contiguous K.
__device__ __forceinline__ v16h load_b_frag(const half_t* base, size_t ldb, int lane) {
    int n  = lane & 15;
    int hf = lane >> 4;
    const half_t* p = base + (size_t)n * ldb;
    v8h lo = *(const v8h*)(p + hf * 8);
    v8h hi = *(const v8h*)(p + 16 + hf * 8);
    v16h b;
#pragma unroll
    for (int i = 0; i < 8; ++i) { b[i] = lo[i]; b[i + 8] = hi[i]; }
    return b;
}

__device__ __forceinline__ v4h cvt4(float4 v) {
    v4h h = { (half_t)v.x, (half_t)v.y, (half_t)v.z, (half_t)v.w };
    return h;
}

// ---------------- Kernel 1: QKV projection GEMM ------------------------------
// qkv = x @ W_qkv^T + b_qkv ; scatter to Q[B,H,T,Dh], K[B,H,T,Dh], Vt[B,H,Dh,T]
__global__ __launch_bounds__(256) void qkv_gemm(
    const float* __restrict__ X, const float* __restrict__ W,
    const float* __restrict__ bias,
    half_t* __restrict__ Qh, half_t* __restrict__ Kh, half_t* __restrict__ Vt)
{
    __shared__ half_t As[2][BM * LDAS];
    __shared__ half_t Bs[2][BN * LDAS];

    const int col0 = blockIdx.x * BN;
    const int row0 = blockIdx.y * BM;
    const int tid  = threadIdx.x;
    const int lane = tid & 31;
    const int wave = tid >> 5;
    const int m    = lane & 15;
    const int hf   = lane >> 4;

    // staging geometry: load i covers row (16*i + sr), col sc..sc+3
    const int sr = tid >> 4;            // 0..15
    const int sc = (tid & 15) * 4;      // 0..60
    const float* pX = X + (size_t)(row0 + sr) * D_MODEL + sc;
    const float* pW = W + (size_t)(col0 + sr) * D_MODEL + sc;
    half_t* sA = &As[0][0] + sr * LDAS + sc;   // buffer 1 via +BM*LDAS
    half_t* sB = &Bs[0][0] + sr * LDAS + sc;

    v8f acc[2][4];
#pragma unroll
    for (int t = 0; t < 2; ++t)
#pragma unroll
        for (int c = 0; c < 4; ++c) acc[t][c] = v8f{};

    float4 ra[16], rb[4];

    // prologue: tile 0 -> regs -> LDS buf 0
#pragma unroll
    for (int i = 0; i < 16; ++i) ra[i] = *(const float4*)(pX + (size_t)i * 16 * D_MODEL);
#pragma unroll
    for (int i = 0; i < 4; ++i)  rb[i] = *(const float4*)(pW + (size_t)i * 16 * D_MODEL);
#pragma unroll
    for (int i = 0; i < 16; ++i) *(v4h*)(sA + i * 16 * LDAS) = cvt4(ra[i]);
#pragma unroll
    for (int i = 0; i < 4; ++i)  *(v4h*)(sB + i * 16 * LDAS) = cvt4(rb[i]);

    int cur = 0;
    for (int it = 0; it < D_MODEL / BK; ++it) {
        __syncthreads();
        const int kn = (it + 1) * BK;
        const bool more = kn < D_MODEL;
        if (more) {  // issue next-tile global loads; pin them before compute
#pragma unroll
            for (int i = 0; i < 16; ++i)
                ra[i] = *(const float4*)(pX + kn + (size_t)i * 16 * D_MODEL);
#pragma unroll
            for (int i = 0; i < 4; ++i)
                rb[i] = *(const float4*)(pW + kn + (size_t)i * 16 * D_MODEL);
        }
        __builtin_amdgcn_sched_barrier(0);   // keep loads issued before WMMAs

        const half_t* At = &As[cur][0];
        const half_t* Bt = &Bs[cur][0];
#pragma unroll
        for (int ks = 0; ks < BK; ks += 32) {
            v16h a0 = load_a_frag(At + (wave * 32) * LDAS + ks, LDAS, lane);
            v16h a1 = load_a_frag(At + (wave * 32 + 16) * LDAS + ks, LDAS, lane);
            v16h b0 = load_b_frag(Bt + 0 * LDAS + ks,  LDAS, lane);
            v16h b1 = load_b_frag(Bt + 16 * LDAS + ks, LDAS, lane);
            v16h b2 = load_b_frag(Bt + 32 * LDAS + ks, LDAS, lane);
            v16h b3 = load_b_frag(Bt + 48 * LDAS + ks, LDAS, lane);
            acc[0][0] = wmma_f16(a0, b0, acc[0][0]);
            acc[0][1] = wmma_f16(a0, b1, acc[0][1]);
            acc[0][2] = wmma_f16(a0, b2, acc[0][2]);
            acc[0][3] = wmma_f16(a0, b3, acc[0][3]);
            acc[1][0] = wmma_f16(a1, b0, acc[1][0]);
            acc[1][1] = wmma_f16(a1, b1, acc[1][1]);
            acc[1][2] = wmma_f16(a1, b2, acc[1][2]);
            acc[1][3] = wmma_f16(a1, b3, acc[1][3]);
        }
        if (more) {
            half_t* dA = sA + (cur ^ 1) * (BM * LDAS);
            half_t* dB = sB + (cur ^ 1) * (BN * LDAS);
#pragma unroll
            for (int i = 0; i < 16; ++i) *(v4h*)(dA + i * 16 * LDAS) = cvt4(ra[i]);
#pragma unroll
            for (int i = 0; i < 4; ++i)  *(v4h*)(dB + i * 16 * LDAS) = cvt4(rb[i]);
        }
        cur ^= 1;
    }

    // epilogue: scatter to Q / K / V^T as f16
#pragma unroll
    for (int t = 0; t < 2; ++t) {
        const int grow0 = row0 + wave * 32 + t * 16;
#pragma unroll
        for (int c = 0; c < 4; ++c) {
            int gcol = col0 + c * 16 + m;          // [0, 3072)
            int kind = gcol >> 10;                 // 0=Q 1=K 2=V
            int dcol = gcol & 1023;
            int h    = dcol >> 6;
            int dh   = dcol & 63;
            float bv = bias[gcol];
#pragma unroll
            for (int r = 0; r < 8; ++r) {
                int grow = grow0 + r + 8 * hf;
                int b    = grow >> 11;             // /SEQ
                int tt   = grow & (SEQ - 1);
                float val = acc[t][c][r] + bv;
                size_t bh = (size_t)(b * NH + h);
                if (kind == 0)      Qh[(bh * SEQ + tt) * DH + dh] = (half_t)val;
                else if (kind == 1) Kh[(bh * SEQ + tt) * DH + dh] = (half_t)val;
                else                Vt[(bh * DH + dh) * SEQ + tt] = (half_t)val;
            }
        }
    }
}

// ---------------- Kernel 2: causal flash attention ---------------------------
// one wave per (b,h, 32-row q tile); K/V fragments shared by both 16-row tiles.
#define PLD 40   // padded P-buffer row stride (halfs): 80B, 16B-aligned
__global__ __launch_bounds__(32) void attn_kernel(
    const half_t* __restrict__ Qh, const half_t* __restrict__ Kh,
    const half_t* __restrict__ Vt, half_t* __restrict__ Oh)
{
    __shared__ half_t Pbuf[32 * PLD];

    const int lane = threadIdx.x & 31;
    const int m    = lane & 15;
    const int hf   = lane >> 4;
    const int q0   = blockIdx.x * 32;
    const int bh   = blockIdx.y;

    const half_t* Qb = Qh + ((size_t)bh * SEQ + q0) * DH;
    const half_t* Kb = Kh + (size_t)bh * SEQ * DH;
    const half_t* Vb = Vt + (size_t)bh * DH * SEQ;

    v16h aq[2][2];
#pragma unroll
    for (int t = 0; t < 2; ++t) {
        aq[t][0] = load_a_frag(Qb + (size_t)(t * 16) * DH,      DH, lane);
        aq[t][1] = load_a_frag(Qb + (size_t)(t * 16) * DH + 32, DH, lane);
    }

    v8f o[2][4];
    float mrow[2][8], lrow[2][8];
#pragma unroll
    for (int t = 0; t < 2; ++t) {
#pragma unroll
        for (int c = 0; c < 4; ++c) o[t][c] = v8f{};
#pragma unroll
        for (int r = 0; r < 8; ++r) { mrow[t][r] = -INFINITY; lrow[t][r] = 0.0f; }
    }

    const int kend = q0 + 32;
    for (int kb = 0; kb < kend; kb += 32) {
        if (kb + 32 < kend) {   // warm next key block (global_prefetch_b8)
            __builtin_prefetch(Kb + (size_t)(kb + 32) * DH, 0, 1);
            __builtin_prefetch(Vb + kb + 32, 0, 1);
        }
        // K fragments for both 16-key tiles (shared by both q row tiles)
        v16h bk00 = load_b_frag(Kb + (size_t)kb * DH,             DH, lane);
        v16h bk01 = load_b_frag(Kb + (size_t)kb * DH + 32,        DH, lane);
        v16h bk10 = load_b_frag(Kb + (size_t)(kb + 16) * DH,      DH, lane);
        v16h bk11 = load_b_frag(Kb + (size_t)(kb + 16) * DH + 32, DH, lane);

        v8f s[2][2];
#pragma unroll
        for (int t = 0; t < 2; ++t) {
            s[t][0] = v8f{}; s[t][1] = v8f{};
            s[t][0] = wmma_f16(aq[t][0], bk00, s[t][0]);
            s[t][0] = wmma_f16(aq[t][1], bk01, s[t][0]);
            s[t][1] = wmma_f16(aq[t][0], bk10, s[t][1]);
            s[t][1] = wmma_f16(aq[t][1], bk11, s[t][1]);
        }

        // issue V fragment loads now; latency hides under softmax VALU below
        v16h bv0 = load_b_frag(Vb + kb,                      SEQ, lane);
        v16h bv1 = load_b_frag(Vb + (size_t)16 * SEQ + kb,   SEQ, lane);
        v16h bv2 = load_b_frag(Vb + (size_t)32 * SEQ + kb,   SEQ, lane);
        v16h bv3 = load_b_frag(Vb + (size_t)48 * SEQ + kb,   SEQ, lane);
        __builtin_amdgcn_sched_barrier(0);   // don't sink V loads to PV site

        const int key0 = kb + m;
        const int key1 = kb + 16 + m;
#pragma unroll
        for (int t = 0; t < 2; ++t) {
#pragma unroll
            for (int r = 0; r < 8; ++r) {
                int q = q0 + t * 16 + r + 8 * hf;
                float v0 = s[t][0][r] * 0.125f;        // 1/sqrt(64)
                float v1 = s[t][1][r] * 0.125f;
                if (key0 > q) v0 = -INFINITY;
                if (key1 > q) v1 = -INFINITY;
                float mx = fmaxf(v0, v1);
                mx = fmaxf(mx, __shfl_xor(mx, 1));
                mx = fmaxf(mx, __shfl_xor(mx, 2));
                mx = fmaxf(mx, __shfl_xor(mx, 4));
                mx = fmaxf(mx, __shfl_xor(mx, 8));
                float mnew  = fmaxf(mrow[t][r], mx);
                float alpha = __expf(mrow[t][r] - mnew);
                float p0 = __expf(v0 - mnew);
                float p1 = __expf(v1 - mnew);
                float rs = p0 + p1;
                rs += __shfl_xor(rs, 1);
                rs += __shfl_xor(rs, 2);
                rs += __shfl_xor(rs, 4);
                rs += __shfl_xor(rs, 8);
                lrow[t][r] = lrow[t][r] * alpha + rs;
                mrow[t][r] = mnew;
                o[t][0][r] *= alpha; o[t][1][r] *= alpha;
                o[t][2][r] *= alpha; o[t][3][r] *= alpha;
                Pbuf[(t * 16 + r + 8 * hf) * PLD + m]      = (half_t)p0;
                Pbuf[(t * 16 + r + 8 * hf) * PLD + 16 + m] = (half_t)p1;
            }
        }
        __syncthreads();   // single-wave WG: LDS ordering fence

        v16h ap0 = load_a_frag(Pbuf,            PLD, lane);
        v16h ap1 = load_a_frag(Pbuf + 16 * PLD, PLD, lane);
        o[0][0] = wmma_f16(ap0, bv0, o[0][0]);
        o[0][1] = wmma_f16(ap0, bv1, o[0][1]);
        o[0][2] = wmma_f16(ap0, bv2, o[0][2]);
        o[0][3] = wmma_f16(ap0, bv3, o[0][3]);
        o[1][0] = wmma_f16(ap1, bv0, o[1][0]);
        o[1][1] = wmma_f16(ap1, bv1, o[1][1]);
        o[1][2] = wmma_f16(ap1, bv2, o[1][2]);
        o[1][3] = wmma_f16(ap1, bv3, o[1][3]);
        __syncthreads();
    }

    const int b = bh >> 4, h = bh & 15;
#pragma unroll
    for (int t = 0; t < 2; ++t) {
#pragma unroll
        for (int r = 0; r < 8; ++r) {
            float inv = 1.0f / lrow[t][r];
            int q = q0 + t * 16 + r + 8 * hf;
            size_t off = ((size_t)b * SEQ + q) * D_MODEL + h * DH;
            Oh[off + m]      = (half_t)(o[t][0][r] * inv);
            Oh[off + 16 + m] = (half_t)(o[t][1][r] * inv);
            Oh[off + 32 + m] = (half_t)(o[t][2][r] * inv);
            Oh[off + 48 + m] = (half_t)(o[t][3][r] * inv);
        }
    }
}

// ---------------- Kernel 3: output projection GEMM ---------------------------
// out = Oh @ W_o^T + b_o   (A already f16, output f32)
__global__ __launch_bounds__(256) void out_gemm(
    const half_t* __restrict__ A, const float* __restrict__ W,
    const float* __restrict__ bias, float* __restrict__ out)
{
    __shared__ half_t As[2][BM * LDAS];
    __shared__ half_t Bs[2][BN * LDAS];

    const int col0 = blockIdx.x * BN;
    const int row0 = blockIdx.y * BM;
    const int tid  = threadIdx.x;
    const int lane = tid & 31;
    const int wave = tid >> 5;
    const int m    = lane & 15;
    const int hf   = lane >> 4;

    // A staging: load i covers row (32*i + tid>>3), col (tid&7)*8 (8 halfs)
    const int arA = tid >> 3;           // 0..31
    const int acA = (tid & 7) * 8;      // 0..56
    const half_t* pA = A + (size_t)(row0 + arA) * D_MODEL + acA;
    half_t* sA = &As[0][0] + arA * LDAS + acA;
    // B staging: load i covers row (16*i + tid>>4), col (tid&15)*4 (4 floats)
    const int srB = tid >> 4;
    const int scB = (tid & 15) * 4;
    const float* pW = W + (size_t)(col0 + srB) * D_MODEL + scB;
    half_t* sB = &Bs[0][0] + srB * LDAS + scB;

    v8f acc[2][4];
#pragma unroll
    for (int t = 0; t < 2; ++t)
#pragma unroll
        for (int c = 0; c < 4; ++c) acc[t][c] = v8f{};

    uint4 ra8[8]; float4 rb[4];

#pragma unroll
    for (int i = 0; i < 8; ++i) ra8[i] = *(const uint4*)(pA + (size_t)i * 32 * D_MODEL);
#pragma unroll
    for (int i = 0; i < 4; ++i) rb[i]  = *(const float4*)(pW + (size_t)i * 16 * D_MODEL);
#pragma unroll
    for (int i = 0; i < 8; ++i) *(uint4*)(sA + i * 32 * LDAS) = ra8[i];
#pragma unroll
    for (int i = 0; i < 4; ++i) *(v4h*)(sB + i * 16 * LDAS) = cvt4(rb[i]);

    int cur = 0;
    for (int it = 0; it < D_MODEL / BK; ++it) {
        __syncthreads();
        const int kn = (it + 1) * BK;
        const bool more = kn < D_MODEL;
        if (more) {
#pragma unroll
            for (int i = 0; i < 8; ++i)
                ra8[i] = *(const uint4*)(pA + kn + (size_t)i * 32 * D_MODEL);
#pragma unroll
            for (int i = 0; i < 4; ++i)
                rb[i] = *(const float4*)(pW + kn + (size_t)i * 16 * D_MODEL);
        }
        __builtin_amdgcn_sched_barrier(0);   // keep loads issued before WMMAs

        const half_t* At = &As[cur][0];
        const half_t* Bt = &Bs[cur][0];
#pragma unroll
        for (int ks = 0; ks < BK; ks += 32) {
            v16h a0 = load_a_frag(At + (wave * 32) * LDAS + ks, LDAS, lane);
            v16h a1 = load_a_frag(At + (wave * 32 + 16) * LDAS + ks, LDAS, lane);
            v16h b0 = load_b_frag(Bt + 0 * LDAS + ks,  LDAS, lane);
            v16h b1 = load_b_frag(Bt + 16 * LDAS + ks, LDAS, lane);
            v16h b2 = load_b_frag(Bt + 32 * LDAS + ks, LDAS, lane);
            v16h b3 = load_b_frag(Bt + 48 * LDAS + ks, LDAS, lane);
            acc[0][0] = wmma_f16(a0, b0, acc[0][0]);
            acc[0][1] = wmma_f16(a0, b1, acc[0][1]);
            acc[0][2] = wmma_f16(a0, b2, acc[0][2]);
            acc[0][3] = wmma_f16(a0, b3, acc[0][3]);
            acc[1][0] = wmma_f16(a1, b0, acc[1][0]);
            acc[1][1] = wmma_f16(a1, b1, acc[1][1]);
            acc[1][2] = wmma_f16(a1, b2, acc[1][2]);
            acc[1][3] = wmma_f16(a1, b3, acc[1][3]);
        }
        if (more) {
            half_t* dA = sA + (cur ^ 1) * (BM * LDAS);
            half_t* dB = sB + (cur ^ 1) * (BN * LDAS);
#pragma unroll
            for (int i = 0; i < 8; ++i) *(uint4*)(dA + i * 32 * LDAS) = ra8[i];
#pragma unroll
            for (int i = 0; i < 4; ++i) *(v4h*)(dB + i * 16 * LDAS) = cvt4(rb[i]);
        }
        cur ^= 1;
    }

#pragma unroll
    for (int t = 0; t < 2; ++t) {
        const int grow0 = row0 + wave * 32 + t * 16;
#pragma unroll
        for (int c = 0; c < 4; ++c) {
            int gcol = col0 + c * 16 + m;
            float bv = bias[gcol];
#pragma unroll
            for (int r = 0; r < 8; ++r) {
                int grow = grow0 + r + 8 * hf;
                out[(size_t)grow * D_MODEL + gcol] = acc[t][c][r] + bv;
            }
        }
    }
}

// ---------------- launcher ---------------------------------------------------
extern "C" void kernel_launch(void* const* d_in, const int* in_sizes, int n_in,
                              void* d_out, int out_size, void* d_ws, size_t ws_size,
                              hipStream_t stream) {
    (void)in_sizes; (void)n_in; (void)out_size; (void)ws_size;
    const float* x     = (const float*)d_in[0];
    const float* W_qkv = (const float*)d_in[1];
    const float* b_qkv = (const float*)d_in[2];
    const float* W_o   = (const float*)d_in[3];
    const float* b_o   = (const float*)d_in[4];
    float* out = (float*)d_out;

    char* ws = (char*)d_ws;
    const size_t seg = (size_t)BATCH * NH * SEQ * DH * sizeof(half_t); // 16 MB
    half_t* Qh = (half_t*)(ws);
    half_t* Kh = (half_t*)(ws + seg);
    half_t* Vt = (half_t*)(ws + 2 * seg);
    half_t* Oh = (half_t*)(ws + 3 * seg);

    dim3 g1((3 * D_MODEL) / BN, MTOT / BM);     // (48, 32)
    qkv_gemm<<<g1, 256, 0, stream>>>(x, W_qkv, b_qkv, Qh, Kh, Vt);

    dim3 g2(SEQ / 32, BATCH * NH);              // (64, 64)
    attn_kernel<<<g2, 32, 0, stream>>>(Qh, Kh, Vt, Oh);

    dim3 g3(D_MODEL / BN, MTOT / BM);           // (16, 32)
    out_gemm<<<g3, 256, 0, stream>>>(Oh, W_o, b_o, out);
}
